// DR2FWL2Conv_81372450390512
// MI455X (gfx1250) — matrix operand order repaired and checked by reading, not compiled, not run on hardware
//
#include <hip/hip_runtime.h>

typedef __attribute__((ext_vector_type(16))) _Float16 v16h;
typedef __attribute__((ext_vector_type(8)))  float    v8f;
typedef __attribute__((ext_vector_type(4)))  float    v4f;

#define GF_ACCUM 1
#define GF_RELU  2

// ---------------------------------------------------------------------------
// Small elementwise kernels
// ---------------------------------------------------------------------------

// convert weight fp32 [k][n] -> f16 TRANSPOSED [n][k] (so B-fragments are
// contiguous 32B runs per lane)
__global__ __launch_bounds__(256) void k_w_to_f16t(const float* __restrict__ s,
                                                   _Float16* __restrict__ d, int n)
{
    int i = blockIdx.x * blockDim.x + threadIdx.x;
    if (i < n) {
        int nn = i >> 6, k = i & 63;
        d[i] = (_Float16)s[k * 64 + nn];
    }
}

__global__ __launch_bounds__(256) void k_zero4(float* __restrict__ p, long n4)
{
    long i = (long)blockIdx.x * blockDim.x + threadIdx.x;
    if (i < n4) {
        v4f z = {0.f, 0.f, 0.f, 0.f};
        ((v4f*)p)[i] = z;
    }
}

// dst[seg[e]] += relu(src[e]) ; thread = (edge, 4-channel chunk)
__global__ __launch_bounds__(256) void k_scatter_relu(const float* __restrict__ src,
                                                      const int* __restrict__ seg,
                                                      float* __restrict__ dst, int rows)
{
    long tid = (long)blockIdx.x * blockDim.x + threadIdx.x;
    if (tid >= (long)rows * 16) return;
    int e = (int)(tid >> 4), q = (int)(tid & 15);
    v4f v = *(const v4f*)(src + (size_t)e * 64 + q * 4);
    float* dp = dst + (size_t)seg[e] * 64 + q * 4;
    atomicAdd(dp + 0, fmaxf(v.x, 0.f));
    atomicAdd(dp + 1, fmaxf(v.y, 0.f));
    atomicAdd(dp + 2, fmaxf(v.z, 0.f));
    atomicAdd(dp + 3, fmaxf(v.w, 0.f));
}

// dst[e] = relu(src[ia[e]] + src[ib[e]])
__global__ __launch_bounds__(256) void k_gather2_relu(const float* __restrict__ src,
                                                      const int* __restrict__ ia,
                                                      const int* __restrict__ ib,
                                                      float* __restrict__ dst, int rows)
{
    long tid = (long)blockIdx.x * blockDim.x + threadIdx.x;
    if (tid >= (long)rows * 16) return;
    int e = (int)(tid >> 4), q = (int)(tid & 15);
    v4f va = *(const v4f*)(src + (size_t)ia[e] * 64 + q * 4);
    v4f vb = *(const v4f*)(src + (size_t)ib[e] * 64 + q * 4);
    v4f r;
    r.x = fmaxf(va.x + vb.x, 0.f);
    r.y = fmaxf(va.y + vb.y, 0.f);
    r.z = fmaxf(va.z + vb.z, 0.f);
    r.w = fmaxf(va.w + vb.w, 0.f);
    *(v4f*)(dst + (size_t)e * 64 + q * 4) = r;
}

// dst[io[t]] += relu(sa[ia[t]] + sb[ib[t]])
__global__ __launch_bounds__(256) void k_tri_scatter(const float* __restrict__ sa,
                                                     const float* __restrict__ sb,
                                                     const int* __restrict__ ia,
                                                     const int* __restrict__ ib,
                                                     const int* __restrict__ io,
                                                     float* __restrict__ dst, int T)
{
    long tid = (long)blockIdx.x * blockDim.x + threadIdx.x;
    if (tid >= (long)T * 16) return;
    int t = (int)(tid >> 4), q = (int)(tid & 15);
    v4f va = *(const v4f*)(sa + (size_t)ia[t] * 64 + q * 4);
    v4f vb = *(const v4f*)(sb + (size_t)ib[t] * 64 + q * 4);
    float* dp = dst + (size_t)io[t] * 64 + q * 4;
    atomicAdd(dp + 0, fmaxf(va.x + vb.x, 0.f));
    atomicAdd(dp + 1, fmaxf(va.y + vb.y, 0.f));
    atomicAdd(dp + 2, fmaxf(va.z + vb.z, 0.f));
    atomicAdd(dp + 3, fmaxf(va.w + vb.w, 0.f));
}

// ---------------------------------------------------------------------------
// WMMA GEMM:  out = op( (A*scale + A2[inv]) @ W + bias [+ out] [+ res] )
// rows % 16 == 0. 256 threads = 8 waves; each wave computes a 16x64 tile via
// 8x v_wmma_f32_16x16x32_f16, stages the tile through its private 4KB LDS
// slice, then does fully-coalesced b128 global read-modify-writes.
// ---------------------------------------------------------------------------
__global__ __launch_bounds__(256)
void k_gemm64(const float* __restrict__ A, const float* __restrict__ A2,
              const int* __restrict__ inv, const float* __restrict__ epsp,
              const _Float16* __restrict__ Wt,   // f16, TRANSPOSED [n][k]
              const float* __restrict__ bias,
              const float* __restrict__ res, float* __restrict__ out,
              int rows, int flags)
{
    __shared__ float lds[8 * 1024];
    const int lane = threadIdx.x & 31;
    const int wv   = threadIdx.x >> 5;
    const int lh   = lane >> 4;      // lane-high half (0/1)
    const int ln   = lane & 15;
    const int ntile = rows >> 4;
    const int wstep = gridDim.x * 8;
    float* smem = lds + wv * 1024;   // wave-private 16x64 tile

    // B fragments: lane holds N = t*16+ln, K = c*32 + lh*16 + e (contiguous e)
    v16h Bf[2][4];
#pragma unroll
    for (int t = 0; t < 4; ++t)
#pragma unroll
        for (int c = 0; c < 2; ++c)
            Bf[c][t] = *(const v16h*)(Wt + (size_t)(t * 16 + ln) * 64 + c * 32 + lh * 16);

    float bcol[4];
#pragma unroll
    for (int t = 0; t < 4; ++t) bcol[t] = bias[t * 16 + ln];

    const float scale    = epsp ? (1.0f + *epsp) : 1.0f;
    const bool  do_accum = (flags & GF_ACCUM) != 0;
    const bool  do_relu  = (flags & GF_RELU) != 0;

    for (int tile = blockIdx.x * 8 + wv; tile < ntile; tile += wstep) {
        const size_t r = (size_t)tile * 16 + ln;   // A row owned by this lane
        const float* ap = A + r * 64;
        const float* ap2 = nullptr;
        if (A2) {
            size_t r2 = inv ? (size_t)inv[r] : r;
            ap2 = A2 + r2 * 64;
        }
        {   // prefetch next tile's A row (clamped in-bounds)
            int tn = tile + wstep; if (tn >= ntile) tn = ntile - 1;
            __builtin_prefetch(A + ((size_t)tn * 16 + ln) * 64, 0, 3);
        }

        // A fragments: lane holds M = ln, K = c*32 + lh*16 + {0..15}
        v16h Af[2];
#pragma unroll
        for (int c = 0; c < 2; ++c) {
            const int o0 = c * 32 + lh * 16;
            v4f g[4];
            g[0] = *(const v4f*)(ap + o0);
            g[1] = *(const v4f*)(ap + o0 + 4);
            g[2] = *(const v4f*)(ap + o0 + 8);
            g[3] = *(const v4f*)(ap + o0 + 12);
            if (ap2) {
                g[0] = g[0] * scale + *(const v4f*)(ap2 + o0);
                g[1] = g[1] * scale + *(const v4f*)(ap2 + o0 + 4);
                g[2] = g[2] * scale + *(const v4f*)(ap2 + o0 + 8);
                g[3] = g[3] * scale + *(const v4f*)(ap2 + o0 + 12);
            }
            v16h a;
#pragma unroll
            for (int j = 0; j < 4; ++j) {
                a[j * 4 + 0] = (_Float16)g[j].x;
                a[j * 4 + 1] = (_Float16)g[j].y;
                a[j * 4 + 2] = (_Float16)g[j].z;
                a[j * 4 + 3] = (_Float16)g[j].w;
            }
            Af[c] = a;
        }

        v8f acc[4];
#pragma unroll
        for (int t = 0; t < 4; ++t) { v8f z = {}; acc[t] = z; }
#pragma unroll
        for (int t = 0; t < 4; ++t) {
            acc[t] = __builtin_amdgcn_wmma_f32_16x16x32_f16(
                false, Af[0], false, Bf[0][t], (short)0, acc[t], false, false);
            acc[t] = __builtin_amdgcn_wmma_f32_16x16x32_f16(
                false, Af[1], false, Bf[1][t], (short)0, acc[t], false, false);
        }

        // stage C/D fragments (+bias) into wave-private LDS tile.
        // C/D layout: VGPR v -> M = v + 8*lh, N = t*16 + ln
#pragma unroll
        for (int t = 0; t < 4; ++t)
#pragma unroll
            for (int v = 0; v < 8; ++v)
                smem[(v + 8 * lh) * 64 + t * 16 + ln] = acc[t][v] + bcol[t];

        // same-wave LDS RAW: DS ops are in-order per wave; fence the compiler.
        asm volatile("" ::: "memory");

        // coalesced readback: lane L, step j -> tile offset j*128 + L*4
        float* gout = out + (size_t)tile * 1024;
        const int lo = lane * 4;
        if (do_accum) {
#pragma unroll
            for (int j = 0; j < 8; ++j) {
                v4f v = *(const v4f*)(smem + j * 128 + lo);
                v4f o = *(const v4f*)(gout + j * 128 + lo);
                *(v4f*)(gout + j * 128 + lo) = v + o;
            }
        } else if (res) {
            const float* rr = res + (size_t)tile * 1024;
#pragma unroll
            for (int j = 0; j < 8; ++j) {
                v4f v = *(const v4f*)(smem + j * 128 + lo);
                v4f o = *(const v4f*)(rr + j * 128 + lo);
                *(v4f*)(gout + j * 128 + lo) = v + o;
            }
        } else if (do_relu) {
#pragma unroll
            for (int j = 0; j < 8; ++j) {
                v4f v = *(const v4f*)(smem + j * 128 + lo);
                v4f o;
                o.x = fmaxf(v.x, 0.f); o.y = fmaxf(v.y, 0.f);
                o.z = fmaxf(v.z, 0.f); o.w = fmaxf(v.w, 0.f);
                *(v4f*)(gout + j * 128 + lo) = o;
            }
        } else {
#pragma unroll
            for (int j = 0; j < 8; ++j)
                *(v4f*)(gout + j * 128 + lo) = *(const v4f*)(smem + j * 128 + lo);
        }
        asm volatile("" ::: "memory");   // next iter's LDS stores stay behind these loads
    }
}

// ---------------------------------------------------------------------------
// Host side
// ---------------------------------------------------------------------------
static void gemm(const float* A, const float* A2, const int* inv, const float* epsp,
                 const _Float16* W, const float* bias, const float* res,
                 float* out, int rows, int flags, hipStream_t s)
{
    int tiles = rows >> 4;
    int blocks = (tiles + 7) / 8;
    if (blocks > 1024) blocks = 1024;
    k_gemm64<<<blocks, 256, 0, s>>>(A, A2, inv, epsp, W, bias, res, out, rows, flags);
}

extern "C" void kernel_launch(void* const* d_in, const int* in_sizes, int n_in,
                              void* d_out, int out_size, void* d_ws, size_t ws_size,
                              hipStream_t stream)
{
    (void)in_sizes; (void)n_in; (void)out_size; (void)ws_size;
    const int N = 10000, E = 320000, T = 500000;

    const float* x0 = (const float*)d_in[0];
    const float* x1 = (const float*)d_in[1];
    const float* x2 = (const float*)d_in[2];
    // weights (converted to f16,transposed): 0 inner,1 lin11,2 lin10,3 lin22,
    // 4 lin20,5 lin21,6 lin12,7 m0W1,8 m0W2,9 m1W1,10 m1W2,11 m2W1,12 m2W2
    const int widx[13] = {3, 5, 7, 9, 11, 13, 15, 17, 19, 21, 23, 25, 27};
    const float* Wsrc[13];
    for (int i = 0; i < 13; ++i) Wsrc[i] = (const float*)d_in[widx[i]];
    const float* b_inner = (const float*)d_in[4];
    const float* b_lin11 = (const float*)d_in[6];
    const float* b_lin10 = (const float*)d_in[8];
    const float* b_lin22 = (const float*)d_in[10];
    const float* b_lin20 = (const float*)d_in[12];
    const float* b_lin21 = (const float*)d_in[14];
    const float* b_lin12 = (const float*)d_in[16];
    const float* b_m0_1 = (const float*)d_in[18], *b_m0_2 = (const float*)d_in[20];
    const float* b_m1_1 = (const float*)d_in[22], *b_m1_2 = (const float*)d_in[24];
    const float* b_m2_1 = (const float*)d_in[26], *b_m2_2 = (const float*)d_in[28];
    const float* eps  = (const float*)d_in[29];
    const int* ei1  = (const int*)d_in[30];
    const int* ei2  = (const int*)d_in[31];
    const int* t111 = (const int*)d_in[32];
    const int* t222 = (const int*)d_in[33];
    const int* t112 = (const int*)d_in[34];
    const int* t221 = (const int*)d_in[35];
    const int* inv1 = (const int*)d_in[36];
    const int* inv2 = (const int*)d_in[37];

    // workspace layout
    _Float16* W16 = (_Float16*)d_ws;
    float* fbase = (float*)((char*)d_ws + 13 * 4096 * sizeof(_Float16)); // 106496 B, 256-aligned
    const size_t NB = (size_t)N * 64, EB = (size_t)E * 64;
    float* inner0 = fbase;            // reused as aggr0
    float* inner1 = inner0 + NB;      // reused as aggr1
    float* inner2 = inner1 + EB;      // reused as aggr2
    float* nodea1 = inner2 + EB;      // ---- zero region start
    float* nodea2 = nodea1 + NB;
    float* T112b  = nodea2 + NB;
    float* T221b  = T112b + EB;
    float* o121   = T221b + EB;
    float* o212   = o121 + EB;        // ---- zero region end
    float* S11    = o212 + EB;        // reused as MLP hidden buffer
    float* S22    = S11 + EB;
    float* aggr0 = inner0; float* aggr1 = inner1; float* aggr2 = inner2;
    float* hbuf  = S11;

    // 1) convert all 13 weight matrices fp32 -> f16 (transposed [n][k])
    for (int i = 0; i < 13; ++i)
        k_w_to_f16t<<<16, 256, 0, stream>>>(Wsrc[i], W16 + (size_t)i * 4096, 4096);

    // 2) zero scatter accumulators
    long zn4 = (long)(2 * NB + 4 * EB) / 4;
    k_zero4<<<(int)((zn4 + 255) / 256), 256, 0, stream>>>(nodea1, zn4);

    // 3) inner = Lin(x, inner_W, inner_b)
    gemm(x0, nullptr, nullptr, nullptr, W16 + 0 * 4096, b_inner, nullptr, inner0, N, 0, stream);
    gemm(x1, nullptr, nullptr, nullptr, W16 + 0 * 4096, b_inner, nullptr, inner1, E, 0, stream);
    gemm(x2, nullptr, nullptr, nullptr, W16 + 0 * 4096, b_inner, nullptr, inner2, E, 0, stream);

    // 4) node accumulators: seg(relu(inner1), s1), seg(relu(inner2), s2)
    int eb = (E * 16 + 255) / 256;
    k_scatter_relu<<<eb, 256, 0, stream>>>(inner1, ei1, nodea1, E);
    k_scatter_relu<<<eb, 256, 0, stream>>>(inner2, ei2, nodea2, E);

    // 5) edge-pair terms (store): relu(inner0[s]+inner0[e])
    k_gather2_relu<<<eb, 256, 0, stream>>>(inner0, ei1, ei1 + E, S11, E);
    k_gather2_relu<<<eb, 256, 0, stream>>>(inner0, ei2, ei2 + E, S22, E);

    // 6) triangle scatters (atomic add)
    int tb = (T * 16 + 255) / 256;
    k_tri_scatter<<<tb, 256, 0, stream>>>(inner1, inner1, t111 + T, t111 + 2 * T, t111, S11, T);
    k_tri_scatter<<<tb, 256, 0, stream>>>(inner2, inner2, t222 + T, t222 + 2 * T, t222, S22, T);
    k_tri_scatter<<<tb, 256, 0, stream>>>(inner1, inner1, t112, t112 + T, t112 + 2 * T, T112b, T);
    k_tri_scatter<<<tb, 256, 0, stream>>>(inner1, inner2, t112 + T, t112 + 2 * T, t112, o121, T);
    k_tri_scatter<<<tb, 256, 0, stream>>>(inner2, inner2, t221, t221 + T, t221 + 2 * T, T221b, T);
    k_tri_scatter<<<tb, 256, 0, stream>>>(inner2, inner1, t221 + T, t221 + 2 * T, t221, o212, T);

    // 7) aggregation linears (inner buffers are dead now; reuse as aggr)
    gemm(S11,  nullptr, nullptr, nullptr, W16 + 1 * 4096, b_lin11, nullptr, aggr1, E, 0, stream);
    gemm(o121, o121,    inv1,    nullptr, W16 + 5 * 4096, b_lin21, nullptr, aggr1, E, GF_ACCUM, stream);
    gemm(T221b,nullptr, nullptr, nullptr, W16 + 3 * 4096, b_lin22, nullptr, aggr1, E, GF_ACCUM, stream);

    gemm(S22,  nullptr, nullptr, nullptr, W16 + 3 * 4096, b_lin22, nullptr, aggr2, E, 0, stream);
    gemm(T112b,nullptr, nullptr, nullptr, W16 + 1 * 4096, b_lin11, nullptr, aggr2, E, GF_ACCUM, stream);
    gemm(o212, o212,    inv2,    nullptr, W16 + 6 * 4096, b_lin12, nullptr, aggr2, E, GF_ACCUM, stream);

    gemm(nodea1, nullptr, nullptr, nullptr, W16 + 2 * 4096, b_lin10, nullptr, aggr0, N, 0, stream);
    gemm(nodea2, nullptr, nullptr, nullptr, W16 + 4 * 4096, b_lin20, nullptr, aggr0, N, GF_ACCUM, stream);

    // 8) GIN-style MLPs with residual: out = MLP(x*(1+eps) + aggr) + x
    float* out0 = (float*)d_out;
    float* out1 = out0 + NB;
    float* out2 = out1 + EB;
    gemm(x0, aggr0, nullptr, eps, W16 + 7 * 4096,  b_m0_1, nullptr, hbuf, N, GF_RELU, stream);
    gemm(hbuf, nullptr, nullptr, nullptr, W16 + 8 * 4096,  b_m0_2, x0, out0, N, 0, stream);
    gemm(x1, aggr1, nullptr, eps, W16 + 9 * 4096,  b_m1_1, nullptr, hbuf, E, GF_RELU, stream);
    gemm(hbuf, nullptr, nullptr, nullptr, W16 + 10 * 4096, b_m1_2, x1, out1, E, 0, stream);
    gemm(x2, aggr2, nullptr, eps, W16 + 11 * 4096, b_m2_1, nullptr, hbuf, E, GF_RELU, stream);
    gemm(hbuf, nullptr, nullptr, nullptr, W16 + 12 * 4096, b_m2_2, x2, out2, E, 0, stream);
}
